// FeatureVolume_48567490183854
// MI455X (gfx1250) — compile-verified, matrix operand order checked by reading, not run on hardware
//
#include <hip/hip_runtime.h>
#include <hip/hip_bf16.h>

typedef __attribute__((ext_vector_type(16))) _Float16 v16h;
typedef __attribute__((ext_vector_type(8)))  float    v8f;

#define RESO 64
#define NP   (RESO*RESO*RESO)      // 262144 points
#define IMW  256
#define IMH  192
#define FC   32                    // feature channels
#define SBLOCKS 512
#define SITERS  16                 // 512 blocks * 16 iters * 8 waves = 65536 groups

// ---------------------------------------------------------------------------
// Stage 1+2: projection, bilinear sampling, masks/weights, WMMA MLP, mean/var
// One wave processes 16 rows (4 views x 4 points) per iteration.
// ---------------------------------------------------------------------------
__global__ __launch_bounds__(256)
void sample_mlp_kernel(const float* __restrict__ feats,   // (4,32,192,256)
                       const float* __restrict__ poses,   // (4,4,4)
                       const float* __restrict__ dprior,  // (4,192,256)
                       const float* __restrict__ w1, const float* __restrict__ b1,
                       const float* __restrict__ w2, const float* __restrict__ b2,
                       const float* __restrict__ w3, const float* __restrict__ b3,
                       float* __restrict__ mv)             // (16,64,64,64)
{
    __shared__ _Float16 sW1[32*32];
    __shared__ _Float16 sW2[32*16];
    __shared__ _Float16 sW3[32*16];
    __shared__ float sB1[32], sB2[16], sB3[8];
    __shared__ _Float16 sA[8][16*32];
    __shared__ _Float16 sH[8][16*32];
    __shared__ float sComp[8][16*8];
    __shared__ float sWD[8][16];
    __shared__ float sWgt[8][16];

    const int tid  = threadIdx.x;
    const int wave = tid >> 5;
    const int lane = tid & 31;

    for (int i = tid; i < 1024; i += 256) sW1[i] = (_Float16)w1[i];
    for (int i = tid; i < 512;  i += 256) sW2[i] = (_Float16)w2[i];
    for (int i = tid; i < 512;  i += 256) {
        int k = i >> 4, n = i & 15;
        sW3[i] = (k < 16 && n < 8) ? (_Float16)w3[k*8+n] : (_Float16)0.f;
    }
    if (tid < 32) sB1[tid] = b1[tid];
    if (tid < 16) sB2[tid] = b2[tid];
    if (tid < 8)  sB3[tid] = b3[tid];
    __syncthreads();

    const int half = lane >> 4;
    const int rr   = lane & 15;
    const int v    = rr >> 2;
    const int pi   = rr & 3;
    const int coln = lane & 15;

    for (int it = 0; it < SITERS; ++it) {
        const int grp = blockIdx.x * (SITERS*8) + it*8 + wave;
        const int n   = grp*4 + pi;
        const int gi = n >> 12, gj = (n >> 6) & 63, gk = n & 63;
        const float x = -1.f + (2.f/63.f)*gi;
        const float y = -1.f + (2.f/63.f)*gj;
        const float z = -1.f + (2.f/63.f)*gk;

        const float* P = poses + v*16;
        const float px = P[0]*x + P[1]*y + P[2]*z  + P[3];
        const float py = P[4]*x + P[5]*y + P[6]*z  + P[7];
        const float pz = P[8]*x + P[9]*y + P[10]*z + P[11];
        const float mask_valid = (pz > 0.f) ? 1.f : 0.f;
        const float invz = 1.f / pz;
        const float gx = px * invz, gy = py * invz;

        const float ix = (gx + 1.f) * 0.5f * (float)IMW - 0.5f;
        const float iy = (gy + 1.f) * 0.5f * (float)IMH - 0.5f;
        const float inm = (ix >= 0.f && ix <= (float)(IMW-1) &&
                           iy >= 0.f && iy <= (float)(IMH-1)) ? 1.f : 0.f;
        const float x0f = floorf(ix), y0f = floorf(iy);
        const int   x0 = (int)x0f,    y0 = (int)y0f;
        const float wx1 = ix - x0f, wy1 = iy - y0f;
        const float wx0 = 1.f - wx1, wy0 = 1.f - wy1;
        auto cl = [](int a, int lo, int hi){ return a < lo ? lo : (a > hi ? hi : a); };
        const int xc0 = cl(x0,   0, IMW-1), xc1 = cl(x0+1, 0, IMW-1);
        const int yc0 = cl(y0,   0, IMH-1), yc1 = cl(y0+1, 0, IMH-1);
        const float v00 = (x0   >= 0 && x0   <= IMW-1 && y0   >= 0 && y0   <= IMH-1) ? 1.f : 0.f;
        const float v10 = (x0+1 >= 0 && x0+1 <= IMW-1 && y0   >= 0 && y0   <= IMH-1) ? 1.f : 0.f;
        const float v01 = (x0   >= 0 && x0   <= IMW-1 && y0+1 >= 0 && y0+1 <= IMH-1) ? 1.f : 0.f;
        const float v11 = (x0+1 >= 0 && x0+1 <= IMW-1 && y0+1 >= 0 && y0+1 <= IMH-1) ? 1.f : 0.f;
        const int o00 = yc0*IMW + xc0, o10 = yc0*IMW + xc1;
        const int o01 = yc1*IMW + xc0, o11 = yc1*IMW + xc1;
        const float w00 = wx0*wy0*v00, w10 = wx1*wy0*v10;
        const float w01 = wx0*wy1*v01, w11 = wx1*wy1*v11;

        const float* dpv = dprior + v*(IMH*IMW);
        float pdep = w00*fmaxf(dpv[o00],0.f) + w10*fmaxf(dpv[o10],0.f)
                   + w01*fmaxf(dpv[o01],0.f) + w11*fmaxf(dpv[o11],0.f);
        pdep *= inm;

        const float projm = (pz > 0.f) ? 1.f : 0.f;
        const float proj  = pz * projm;
        float zf = -P[8] - P[9] - P[10] + P[11];
        float zl =  P[8] + P[9] + P[10] + P[11];
        zf = (zf > 0.f) ? zf : 0.f;
        zl = (zl > 0.f) ? zl : 0.f;
        const float unit = (zl - zf) * (1.f/63.f);
        const float dmask = ((fabsf(pdep - proj) < 50.f*unit) ? 1.f : 0.f) * inm * projm;
        const float m = inm * mask_valid;

        const float msum = __shfl(m, pi, 32) + __shfl(m, pi+4, 32)
                         + __shfl(m, pi+8, 32) + __shfl(m, pi+12, 32);
        const float dsum = __shfl(dmask, pi, 32) + __shfl(dmask, pi+4, 32)
                         + __shfl(dmask, pi+8, 32) + __shfl(dmask, pi+12, 32);
        const float wgt = m / (msum + 1e-8f);
        const float dwg = dmask / (dsum + 1e-8f);
        if (lane < 16) { sWD[wave][lane] = wgt * dwg; sWgt[wave][lane] = wgt; }

        const float* fv = feats + v*(FC*IMH*IMW);
        const int cbase = half * 16;
        #pragma unroll 4
        for (int c = 0; c < 16; ++c) {
            const float* fcp = fv + (cbase + c)*(IMH*IMW);
            float val = w00*fcp[o00] + w10*fcp[o10] + w01*fcp[o01] + w11*fcp[o11];
            sA[wave][rr*32 + cbase + c] = (_Float16)(val * dwg);
        }
        __syncthreads();

        v16h afr;
        #pragma unroll
        for (int r = 0; r < 8; ++r) {
            const int k0 = ((r < 4) ? 0 : 16) + half*8 + (r & 3)*2;
            afr[2*r]   = sA[wave][rr*32 + k0];
            afr[2*r+1] = sA[wave][rr*32 + k0 + 1];
        }
        auto loadB = [&](const _Float16* Wm, int ncols, int nbase) -> v16h {
            v16h bfr;
            const int col = nbase + coln;
            #pragma unroll
            for (int r = 0; r < 8; ++r) {
                const int k0 = ((r < 4) ? 0 : 16) + half*8 + (r & 3)*2;
                bfr[2*r]   = Wm[k0*ncols + col];
                bfr[2*r+1] = Wm[(k0+1)*ncols + col];
            }
            return bfr;
        };

        // Layer 1: 32 -> 32 (two N tiles)
        v16h b0 = loadB(sW1, 32, 0);
        v16h b1f = loadB(sW1, 32, 16);
        v8f d0 = {}, d1 = {};
        d0 = __builtin_amdgcn_wmma_f32_16x16x32_f16(false, afr, false, b0,  (short)0, d0, false, false);
        d1 = __builtin_amdgcn_wmma_f32_16x16x32_f16(false, afr, false, b1f, (short)0, d1, false, false);
        #pragma unroll
        for (int e = 0; e < 8; ++e) {
            const int Mr = e + half*8;
            sH[wave][Mr*32 + coln]      = (_Float16)fmaxf(d0[e] + sB1[coln],      0.f);
            sH[wave][Mr*32 + 16 + coln] = (_Float16)fmaxf(d1[e] + sB1[16 + coln], 0.f);
        }
        __syncthreads();

        // Layer 2: 32 -> 16
        v16h a2;
        #pragma unroll
        for (int r = 0; r < 8; ++r) {
            const int k0 = ((r < 4) ? 0 : 16) + half*8 + (r & 3)*2;
            a2[2*r]   = sH[wave][rr*32 + k0];
            a2[2*r+1] = sH[wave][rr*32 + k0 + 1];
        }
        v16h b2f = loadB(sW2, 16, 0);
        v8f d2 = {};
        d2 = __builtin_amdgcn_wmma_f32_16x16x32_f16(false, a2, false, b2f, (short)0, d2, false, false);
        #pragma unroll
        for (int e = 0; e < 8; ++e) {
            const int Mr = e + half*8;
            sA[wave][Mr*32 + coln]      = (_Float16)fmaxf(d2[e] + sB2[coln], 0.f);
            sA[wave][Mr*32 + 16 + coln] = (_Float16)0.f;   // K pad for layer 3
        }
        __syncthreads();

        // Layer 3: 16 -> 8 (K padded to 32, N padded to 16)
        v16h a3;
        #pragma unroll
        for (int r = 0; r < 8; ++r) {
            const int k0 = ((r < 4) ? 0 : 16) + half*8 + (r & 3)*2;
            a3[2*r]   = sA[wave][rr*32 + k0];
            a3[2*r+1] = sA[wave][rr*32 + k0 + 1];
        }
        v16h b3f = loadB(sW3, 16, 0);
        v8f d3 = {};
        d3 = __builtin_amdgcn_wmma_f32_16x16x32_f16(false, a3, false, b3f, (short)0, d3, false, false);
        if (coln < 8) {
            #pragma unroll
            for (int e = 0; e < 8; ++e) {
                const int Mr = e + half*8;
                sComp[wave][Mr*8 + coln] = d3[e] + sB3[coln];
            }
        }
        __syncthreads();

        {
            const int pp = lane >> 3, ch = lane & 7;
            float mean = 0.f;
            #pragma unroll
            for (int vv = 0; vv < 4; ++vv)
                mean += sComp[wave][(vv*4+pp)*8 + ch] * sWD[wave][vv*4+pp];
            float var = 0.f;
            #pragma unroll
            for (int vv = 0; vv < 4; ++vv) {
                const float dlt = sComp[wave][(vv*4+pp)*8 + ch] - mean;
                var += sWgt[wave][vv*4+pp] * dlt * dlt;
            }
            const int n2 = grp*4 + pp;
            const int ii = n2 >> 12, jj = (n2 >> 6) & 63, kk = n2 & 63;
            const int off = kk*4096 + jj*64 + ii;
            mv[ch*NP + off]       = mean;
            mv[(8 + ch)*NP + off] = var;
        }
        __syncthreads();
    }
}

// ---------------------------------------------------------------------------
// Stage 3: implicit-GEMM WMMA conv / transposed-conv.
// M tile = 16 output positions (one wave), N tiles = 16 channels each (<=4),
// K = Cin*27 in chunks of 32, f16 operands, f32 accumulation in C operand.
// ---------------------------------------------------------------------------
__global__ __launch_bounds__(256)
void conv3d_wmma(const float* __restrict__ in, const float* __restrict__ w,
                 float* __restrict__ raw,
                 int Cin, int Co, int Din, int Dout, int stride, int deconv)
{
    __shared__ _Float16 sAc[8][512];    // per-wave A tile: 16 rows x 32 K
    __shared__ _Float16 sBc[4*512];     // shared B slab: 4 N-tiles x 32 K x 16 N

    const int tid  = threadIdx.x;
    const int wave = tid >> 5;
    const int lane = tid & 31;
    const int half = lane >> 4;
    const int coln = lane & 15;

    const int spatial = Dout*Dout*Dout;
    const int tile = blockIdx.x*8 + wave;
    const int Ktot = Cin*27;
    const int KC = (Ktot + 31) >> 5;
    const int NT = (Co + 15) >> 4;

    // each lane owns one output row of the M tile and half of the K chunk
    const int mrow = lane >> 1;
    const int kkb  = (lane & 1) * 16;
    const int midx = tile*16 + mrow;
    const int zz   = midx / (Dout*Dout);
    const int rem  = midx - zz*Dout*Dout;
    const int yy   = rem / Dout, xx = rem - yy*Dout;
    const int din2 = Din*Din;

    v8f acc[4] = {{}, {}, {}, {}};

    for (int kc = 0; kc < KC; ++kc) {
        __syncthreads();
        // ---- stage B (weights) for all N tiles, block-cooperative
        for (int i = tid; i < NT*512; i += 256) {
            const int nt = i >> 9;
            const int r  = i & 511;
            const int kk = r >> 4, nn = r & 15;
            const int k  = kc*32 + kk;
            const int co = nt*16 + nn;
            float val = 0.f;
            if (k < Ktot && co < Co) {
                const int ci = k / 27, tap = k - ci*27;
                val = deconv ? w[(ci*Co + co)*27 + (26 - tap)]
                             : w[(co*Cin + ci)*27 + tap];
            }
            sBc[i] = (_Float16)val;
        }
        // ---- stage A (im2col gather), 16 values per lane
        #pragma unroll 4
        for (int e = 0; e < 16; ++e) {
            const int kk = kkb + e;
            const int k  = kc*32 + kk;
            float val = 0.f;
            if (k < Ktot) {
                const int ci = k / 27, tap = k - ci*27;
                const int dz = tap / 9, r9 = tap - dz*9;
                const int dy = r9 / 3, dx = r9 - dy*3;
                if (!deconv) {
                    const int iz = zz*stride + dz - 1;
                    const int iy = yy*stride + dy - 1;
                    const int ixx = xx*stride + dx - 1;
                    if (iz >= 0 && iz < Din && iy >= 0 && iy < Din && ixx >= 0 && ixx < Din)
                        val = in[ci*Din*din2 + iz*din2 + iy*Din + ixx];
                } else {
                    const int pz = zz + dz, py = yy + dy, pxv = xx + dx;
                    if ((pz & 1) && (py & 1) && (pxv & 1)) {
                        const int iz = (pz-1) >> 1, iy = (py-1) >> 1, ixx = (pxv-1) >> 1;
                        if (iz < Din && iy < Din && ixx < Din)
                            val = in[ci*Din*din2 + iz*din2 + iy*Din + ixx];
                    }
                }
            }
            sAc[wave][mrow*32 + kk] = (_Float16)val;
        }
        __syncthreads();
        // ---- fragments + WMMA accumulate
        v16h afr;
        #pragma unroll
        for (int r = 0; r < 8; ++r) {
            const int k0 = ((r < 4) ? 0 : 16) + half*8 + (r & 3)*2;
            afr[2*r]   = sAc[wave][coln*32 + k0];
            afr[2*r+1] = sAc[wave][coln*32 + k0 + 1];
        }
        #pragma unroll
        for (int nt = 0; nt < 4; ++nt) {
            if (nt < NT) {
                v16h bfr;
                #pragma unroll
                for (int r = 0; r < 8; ++r) {
                    const int k0 = ((r < 4) ? 0 : 16) + half*8 + (r & 3)*2;
                    bfr[2*r]   = sBc[nt*512 + k0*16 + coln];
                    bfr[2*r+1] = sBc[nt*512 + (k0+1)*16 + coln];
                }
                acc[nt] = __builtin_amdgcn_wmma_f32_16x16x32_f16(
                    false, afr, false, bfr, (short)0, acc[nt], false, false);
            }
        }
    }
    // ---- store raw conv output
    #pragma unroll
    for (int nt = 0; nt < 4; ++nt) {
        if (nt < NT) {
            const int co = nt*16 + coln;
            if (co < Co) {
                float* rp = raw + (size_t)co*spatial + tile*16 + half*8;
                #pragma unroll
                for (int e = 0; e < 8; ++e) rp[e] = acc[nt][e];
            }
        }
    }
}

// ---------------------------------------------------------------------------
// BatchNorm statistics / finalize / apply
// ---------------------------------------------------------------------------
__global__ __launch_bounds__(256)
void bn_stats(const float* __restrict__ raw, float* __restrict__ stats, int spatial)
{
    const int co = blockIdx.y;
    const int idx = blockIdx.x*256 + threadIdx.x;
    const float val = (idx < spatial) ? raw[(size_t)co*spatial + idx] : 0.f;
    __shared__ float rs[256], rq[256];
    rs[threadIdx.x] = val;
    rq[threadIdx.x] = val*val;
    __syncthreads();
    for (int s = 128; s > 0; s >>= 1) {
        if (threadIdx.x < s) {
            rs[threadIdx.x] += rs[threadIdx.x + s];
            rq[threadIdx.x] += rq[threadIdx.x + s];
        }
        __syncthreads();
    }
    if (threadIdx.x == 0) {
        atomicAdd(&stats[co],      rs[0]);
        atomicAdd(&stats[64 + co], rq[0]);
    }
}

__global__ void zero128(float* s) { s[threadIdx.x] = 0.f; }

__global__ void bn_finalize(float* stats, const float* __restrict__ g,
                            const float* __restrict__ b, int count)
{
    const int c = threadIdx.x;
    const float inv = 1.f / (float)count;
    const float mean = stats[c] * inv;
    const float var  = stats[64 + c] * inv - mean*mean;
    const float sc = g[c] * rsqrtf(var + 1e-5f);
    stats[c]      = sc;
    stats[64 + c] = b[c] - mean*sc;
}

__global__ __launch_bounds__(256)
void bn_relu_add(const float* __restrict__ raw, const float* __restrict__ stats,
                 const float* __restrict__ skip, float* __restrict__ out,
                 int spatial, int total)
{
    const int idx = blockIdx.x*256 + threadIdx.x;
    if (idx >= total) return;
    const int co = idx / spatial;
    float v = fmaxf(raw[idx]*stats[co] + stats[64 + co], 0.f);
    if (skip) v += skip[idx];
    out[idx] = v;
}

// ---------------------------------------------------------------------------
extern "C" void kernel_launch(void* const* d_in, const int* in_sizes, int n_in,
                              void* d_out, int out_size, void* d_ws, size_t ws_size,
                              hipStream_t stream) {
    const float* feats  = (const float*)d_in[0];
    const float* poses  = (const float*)d_in[1];
    const float* dprior = (const float*)d_in[2];
    const float* w1 = (const float*)d_in[3];
    const float* b1 = (const float*)d_in[4];
    const float* w2 = (const float*)d_in[5];
    const float* b2 = (const float*)d_in[6];
    const float* w3 = (const float*)d_in[7];
    const float* b3 = (const float*)d_in[8];

    float* ws    = (float*)d_ws;
    float* mv    = ws;                                 // 16ch * 64^3  = 4M floats
    float* raw   = mv    + (size_t)16*NP;              // up to 2M floats
    float* skip0 = raw   + (size_t)8*NP;               // 8 * 64^3     = 2M
    float* skip2 = skip0 + (size_t)8*NP;               // 16 * 32^3    = 512K
    float* skip4 = skip2 + (size_t)16*32768;           // 32 * 16^3    = 128K
    float* ping  = skip4 + (size_t)32*4096;            // 512K
    float* pong  = ping  + (size_t)16*32768;           // 512K
    float* stats = pong  + (size_t)16*32768;           // 128

    sample_mlp_kernel<<<SBLOCKS, 256, 0, stream>>>(feats, poses, dprior,
                                                   w1, b1, w2, b2, w3, b3, mv);

    auto CBR = [&](const float* inb, int wi, float* normOut, const float* skip,
                   int Cin, int Co, int Din, int Dout, int stride, int deconv) {
        zero128<<<1, 128, 0, stream>>>(stats);
        const int spatial = Dout*Dout*Dout;
        conv3d_wmma<<<spatial/128, 256, 0, stream>>>(inb, (const float*)d_in[wi], raw,
                                                     Cin, Co, Din, Dout, stride, deconv);
        dim3 gs((spatial + 255)/256, Co);
        bn_stats<<<gs, 256, 0, stream>>>(raw, stats, spatial);
        bn_finalize<<<1, Co, 0, stream>>>(stats, (const float*)d_in[wi+1],
                                          (const float*)d_in[wi+2], spatial);
        const int tot = Co*spatial;
        bn_relu_add<<<(tot + 255)/256, 256, 0, stream>>>(raw, stats, skip, normOut,
                                                         spatial, tot);
    };

    CBR(mv,    9,  skip0, nullptr, 16,  8, 64, 64, 1, 0);   // c0
    CBR(skip0, 12, ping,  nullptr,  8, 16, 64, 32, 2, 0);   // c1
    CBR(ping,  15, skip2, nullptr, 16, 16, 32, 32, 1, 0);   // c2
    CBR(skip2, 18, ping,  nullptr, 16, 32, 32, 16, 2, 0);   // c3
    CBR(ping,  21, skip4, nullptr, 32, 32, 16, 16, 1, 0);   // c4
    CBR(skip4, 24, ping,  nullptr, 32, 64, 16,  8, 2, 0);   // c5
    CBR(ping,  27, pong,  nullptr, 64, 64,  8,  8, 1, 0);   // c6
    CBR(pong,  30, ping,  skip4,   64, 32,  8, 16, 1, 1);   // d7  (+c4)
    CBR(ping,  33, pong,  skip2,   32, 16, 16, 32, 1, 1);   // d9  (+c2)
    CBR(pong,  36, (float*)d_out, skip0, 16, 8, 32, 64, 1, 1); // d11 (+c0)
}
